// HeteroGNN_57492432224469
// MI455X (gfx1250) — compile-verified, matrix operand order checked by reading, not run on hardware
//
#include <hip/hip_runtime.h>
#include <hip/hip_bf16.h>

// ---------- problem constants ----------
#define N_A   100000
#define N_B   100000
#define NE    500000
#define HD    128
#define NH    (100000 * 128)   // elements of one node-feature matrix

typedef __attribute__((ext_vector_type(16))) __bf16      v16bf;
typedef __attribute__((ext_vector_type(8)))  float       v8f;
typedef __attribute__((ext_vector_type(8)))  unsigned    v8u;

// round-to-nearest-even f32 -> bf16, packed pair into one dword
__device__ __forceinline__ unsigned pk_bf16(float x, float y) {
    unsigned ux = __float_as_uint(x), uy = __float_as_uint(y);
    ux = (ux + 0x7FFFu + ((ux >> 16) & 1u)) >> 16;
    uy = (uy + 0x7FFFu + ((uy >> 16) & 1u)) >> 16;
    return (ux & 0xFFFFu) | (uy << 16);
}

// ---------- weight swizzle: row-major f32 W[K][N] -> WMMA bf16 B-operand tiles ----------
// storage: dword index = ((tn*(K/32)+tk)*256 + lane*8 + v)
// lane l holds column n = tn*16 + (l&15); VGPR v holds K = tk*32 + (v>=4?16:0) + (l>=16?8:0) + (v&3)*2 + {0,1}
__global__ void k_swizzle(const float* __restrict__ W, unsigned* __restrict__ dst,
                          int K, int N) {
    int id = blockIdx.x * 256 + threadIdx.x;
    int total = (K >> 5) * (N >> 4) * 256;
    if (id >= total) return;
    int tile = id >> 8, r = id & 255;
    int lane = r >> 3, v = r & 7;
    int ktiles = K >> 5;
    int tn = tile / ktiles, tk = tile % ktiles;
    int n  = (tn << 4) + (lane & 15);
    int kb = (tk << 5) + ((v >> 2) << 4) + ((lane >> 4) << 3) + ((v & 3) << 1);
    dst[id] = pk_bf16(W[(size_t)kb * N + n], W[(size_t)(kb + 1) * N + n]);
}

// ---------- WMMA GEMM: Out[M x (NT*16)] = act(A[M x (KT*32)] @ Wswz + bias) ----------
// B (swizzled weights, NT*KT*1KB) staged in LDS once per block, shared by 8 waves.
// Per k-step: issue A global loads, then ALL NT B-tile ds_loads into a register
// array, pack A while loads are in flight, then NT back-to-back WMMAs.
template <int NT, int KT, bool ABF16, bool BIAS, bool RELU>
__global__ __launch_bounds__(256) void k_gemm(const void* __restrict__ Ain,
                                              const unsigned* __restrict__ Bswz,
                                              const float* __restrict__ bias,
                                              float* __restrict__ Out, int M) {
    __shared__ __align__(16) unsigned smem[NT * KT * 256];

    // ---- stage all B tiles to LDS (whole block participates, then barrier) ----
    {
        const uint4* g = (const uint4*)Bswz;
        uint4* s = (uint4*)smem;
#pragma unroll
        for (int i = threadIdx.x; i < NT * KT * 64; i += 256) s[i] = g[i];
    }
    __syncthreads();

    const int lane    = threadIdx.x & 31;
    const int wave    = threadIdx.x >> 5;
    const int rowbase = blockIdx.x * 128 + wave * 16;
    if (rowbase >= M) return;                // uniform per wave (after barrier)
    int row = rowbase + (lane & 15);
    if (row >= M) row = M - 1;               // keep EXEC full; stores guarded below

    v8f acc[NT];
    v8f zero = {0.f, 0.f, 0.f, 0.f, 0.f, 0.f, 0.f, 0.f};
#pragma unroll
    for (int t = 0; t < NT; ++t) acc[t] = zero;

    const float*    arf = (const float*)Ain    + (size_t)row * (KT * 32) + ((lane >> 4) << 3);
    const unsigned* arb = (const unsigned*)Ain + (size_t)row * (KT * 16) + ((lane >> 4) << 2);
    const unsigned* sb  = smem + (lane << 3);

#pragma unroll
    for (int tk = 0; tk < KT; ++tk) {
        // 1) issue A global loads first (longest latency)
        uint4  c0, c1;                 // bf16 path
        float4 f0, f1, g0, g1;         // f32 path
        if (ABF16) {
            const uint4* p = (const uint4*)(arb + (tk << 4));
            c0 = p[0]; c1 = p[2];      // K base+0..7 , base+16..23
        } else {
            const float4* p = (const float4*)(arf + (tk << 5));
            f0 = p[0]; f1 = p[1];      // K base+0..7
            g0 = p[4]; g1 = p[5];      // K base+16..23
        }
        // 2) issue ALL B-tile LDS loads into a register array
        v8u bu[NT];
#pragma unroll
        for (int tn = 0; tn < NT; ++tn)
            bu[tn] = *(const v8u*)(sb + ((tn * KT + tk) << 8));
        // 3) build A operand (pack VALU overlaps in-flight loads / co-executes with XDL)
        v8u au;
        if (ABF16) {
            au[0] = c0.x; au[1] = c0.y; au[2] = c0.z; au[3] = c0.w;
            au[4] = c1.x; au[5] = c1.y; au[6] = c1.z; au[7] = c1.w;
        } else {
            au[0] = pk_bf16(f0.x, f0.y); au[1] = pk_bf16(f0.z, f0.w);
            au[2] = pk_bf16(f1.x, f1.y); au[3] = pk_bf16(f1.z, f1.w);
            au[4] = pk_bf16(g0.x, g0.y); au[5] = pk_bf16(g0.z, g0.w);
            au[6] = pk_bf16(g1.x, g1.y); au[7] = pk_bf16(g1.z, g1.w);
        }
        v16bf a = __builtin_bit_cast(v16bf, au);
        // 4) NT back-to-back WMMAs (independent accumulators)
#pragma unroll
        for (int tn = 0; tn < NT; ++tn) {
            v16bf b = __builtin_bit_cast(v16bf, bu[tn]);
            acc[tn] = __builtin_amdgcn_wmma_f32_16x16x32_bf16(
                false, a, false, b, (short)0, acc[tn], false, false);
        }
    }

    const int col   = lane & 15;
    const int rhalf = (lane >> 4) << 3;   // 0 or 8
    const int Nst   = NT * 16;
#pragma unroll
    for (int tn = 0; tn < NT; ++tn) {
        float bv = BIAS ? bias[tn * 16 + col] : 0.0f;
#pragma unroll
        for (int r = 0; r < 8; ++r) {
            int orow = rowbase + rhalf + r;
            if (orow < M) {
                float v = acc[tn][r] + bv;
                if (RELU) v = fmaxf(v, 0.0f);
                Out[(size_t)orow * Nst + tn * 16 + col] = v;
            }
        }
    }
}

// ---------- misc elementwise / graph kernels ----------
__global__ void k_fill0(float* p, int n) {
    int i = blockIdx.x * 256 + threadIdx.x;
    if (i < n) p[i] = 0.0f;
}
__global__ void k_count(const int* __restrict__ dst, float* __restrict__ cnt, int ne) {
    int i = blockIdx.x * 256 + threadIdx.x;
    if (i < ne) atomicAdd(&cnt[dst[i]], 1.0f);
}
__global__ void k_inv(float* c, int n) {
    int i = blockIdx.x * 256 + threadIdx.x;
    if (i < n) c[i] = 1.0f / fmaxf(c[i], 1.0f);
}
// out[dst] += msg[src] * inv[dst]   (pre-scaled scatter-mean), 32 threads/edge
__global__ void k_scatter(const float* __restrict__ msg, const int* __restrict__ src,
                          const int* __restrict__ dst, const float* __restrict__ inv,
                          float* __restrict__ out, int ne) {
    int tid = blockIdx.x * 256 + threadIdx.x;
    int e = tid >> 5;
    if (e >= ne) return;
    int c = (tid & 31) << 2;
    int s = src[e], d = dst[e];
    float w = inv[d];
    float4 m = *(const float4*)(msg + (size_t)s * HD + c);
    float* o = out + (size_t)d * HD + c;
    atomicAdd(o + 0, m.x * w);
    atomicAdd(o + 1, m.y * w);
    atomicAdd(o + 2, m.z * w);
    atomicAdd(o + 3, m.w * w);
}
// f32 -> packed bf16 mirror
__global__ void k_cvt(const float4* __restrict__ src, uint2* __restrict__ dst, int n4) {
    int i = blockIdx.x * 256 + threadIdx.x;
    if (i >= n4) return;
    float4 f = src[i];
    dst[i] = make_uint2(pk_bf16(f.x, f.y), pk_bf16(f.z, f.w));
}
// h = relu(out) + h ; also refresh packed-bf16 mirror of h
__global__ void k_combine(float4* __restrict__ h, const float4* __restrict__ o,
                          uint2* __restrict__ hbf, int n4) {
    int i = blockIdx.x * 256 + threadIdx.x;
    if (i >= n4) return;
    float4 a = o[i], b = h[i];
    b.x += fmaxf(a.x, 0.f); b.y += fmaxf(a.y, 0.f);
    b.z += fmaxf(a.z, 0.f); b.w += fmaxf(a.w, 0.f);
    h[i] = b;
    hbf[i] = make_uint2(pk_bf16(b.x, b.y), pk_bf16(b.z, b.w));
}
// pred[i] = z[i,:] . w + b   (z is M x 64)
__global__ void k_head2(const float* __restrict__ z, const float* __restrict__ w,
                        const float* __restrict__ b, float* __restrict__ pred, int n) {
    int i = blockIdx.x * 256 + threadIdx.x;
    if (i >= n) return;
    const float4* zp = (const float4*)(z + (size_t)i * 64);
    const float4* wp = (const float4*)w;
    float s = 0.f;
#pragma unroll
    for (int j = 0; j < 16; ++j) {
        float4 a = zp[j], c = wp[j];
        s += a.x * c.x + a.y * c.y + a.z * c.z + a.w * c.w;
    }
    pred[i] = s + b[0];
}

// ---------- swizzled-weight dword offsets ----------
#define DW_128x128 8192      // K*N/2
#define OFF_IPA    0                               // 256x128 -> 16384
#define OFF_IPB    16384                           // 128x128
#define OFF_L0     24576                           // per-layer: aa,ab,ba,selfA,selfB (5 x 8192)
#define L_STRIDE   (5 * DW_128x128)
#define OFF_HA     (OFF_L0 + 2 * L_STRIDE)         // 128x64 -> 4096
#define OFF_HB     (OFF_HA + 4096)

extern "C" void kernel_launch(void* const* d_in, const int* in_sizes, int n_in,
                              void* d_out, int out_size, void* d_ws, size_t ws_size,
                              hipStream_t stream) {
    (void)in_sizes; (void)n_in; (void)out_size; (void)ws_size;

    // ---- inputs (setup_inputs order; params leaves in pytree (sorted-key) order) ----
    const float* xA    = (const float*)d_in[0];
    const float* xB    = (const float*)d_in[1];
    const int*   ei_ab = (const int*)d_in[2];
    const int*   ei_ba = (const int*)d_in[3];
    const int*   ei_aa = (const int*)d_in[4];
    const float* hA_l1W = (const float*)d_in[5];
    const float* hA_l1b = (const float*)d_in[6];
    const float* hA_l2W = (const float*)d_in[7];
    const float* hA_l2b = (const float*)d_in[8];
    const float* hB_l1W = (const float*)d_in[9];
    const float* hB_l1b = (const float*)d_in[10];
    const float* hB_l2W = (const float*)d_in[11];
    const float* hB_l2b = (const float*)d_in[12];
    const float* ipA_W = (const float*)d_in[13];
    const float* ipA_b = (const float*)d_in[14];
    const float* ipB_W = (const float*)d_in[15];
    const float* ipB_b = (const float*)d_in[16];
    const float* eaaW[2], *eabW[2], *ebaW[2], *sAW[2], *sAb[2], *sBW[2], *sBb[2];
    for (int l = 0; l < 2; ++l) {
        int base = 17 + 7 * l;
        eaaW[l] = (const float*)d_in[base + 0];
        eabW[l] = (const float*)d_in[base + 1];
        ebaW[l] = (const float*)d_in[base + 2];
        sAW[l]  = (const float*)d_in[base + 3];
        sAb[l]  = (const float*)d_in[base + 4];
        sBW[l]  = (const float*)d_in[base + 5];
        sBb[l]  = (const float*)d_in[base + 6];
    }
    float* pred = (float*)d_out;

    // ---- workspace layout ----
    float* ws      = (float*)d_ws;
    float* hA      = ws;
    float* hB      = ws + (size_t)NH;
    float* outA    = ws + (size_t)2 * NH;
    float* outB    = ws + (size_t)3 * NH;
    float* msg     = ws + (size_t)4 * NH;          // also head intermediate z
    float* cntAB   = ws + (size_t)5 * NH;          // inv mean counts
    float* cntBA   = cntAB + N_B;
    float* cntAA   = cntBA + N_A;
    unsigned* hAbf = (unsigned*)(cntAA + N_A);     // packed bf16 mirrors (NH/2 dwords each)
    unsigned* hBbf = hAbf + (size_t)NH / 2;
    unsigned* swz  = hBbf + (size_t)NH / 2;

    // ---- 1) swizzle all weights into WMMA B-operand layout (bf16 packed) ----
    auto swl = [&](const float* W, int off, int K, int N) {
        int total = (K * N) / 2;
        k_swizzle<<<(total + 255) / 256, 256, 0, stream>>>(W, swz + off, K, N);
    };
    swl(ipA_W, OFF_IPA, 256, 128);
    swl(ipB_W, OFF_IPB, 128, 128);
    for (int l = 0; l < 2; ++l) {
        int b = OFF_L0 + l * L_STRIDE;
        swl(eaaW[l], b,                  128, 128);
        swl(eabW[l], b + DW_128x128,     128, 128);
        swl(ebaW[l], b + 2 * DW_128x128, 128, 128);
        swl(sAW[l],  b + 3 * DW_128x128, 128, 128);
        swl(sBW[l],  b + 4 * DW_128x128, 128, 128);
    }
    swl(hA_l1W, OFF_HA, 128, 64);
    swl(hB_l1W, OFF_HB, 128, 64);

    // ---- 2) per-edge-type inverse mean counts (constant across layers) ----
    int gn = (100000 + 255) / 256, ge = (NE + 255) / 256;
    k_fill0<<<gn, 256, 0, stream>>>(cntAB, N_B);
    k_fill0<<<gn, 256, 0, stream>>>(cntBA, N_A);
    k_fill0<<<gn, 256, 0, stream>>>(cntAA, N_A);
    k_count<<<ge, 256, 0, stream>>>(ei_ab + NE, cntAB, NE);
    k_count<<<ge, 256, 0, stream>>>(ei_ba + NE, cntBA, NE);
    k_count<<<ge, 256, 0, stream>>>(ei_aa + NE, cntAA, NE);
    k_inv<<<gn, 256, 0, stream>>>(cntAB, N_B);
    k_inv<<<gn, 256, 0, stream>>>(cntBA, N_A);
    k_inv<<<gn, 256, 0, stream>>>(cntAA, N_A);

    // ---- 3) input projection + relu, then build bf16 mirrors ----
    int gm = (100000 + 127) / 128;  // 782 blocks, 8 waves x 16 rows each
    int gc = (NH / 4 + 255) / 256;  // float4-granularity elementwise
    k_gemm<8, 8, false, true, true><<<gm, 256, 0, stream>>>(xA, swz + OFF_IPA, ipA_b, hA, N_A);
    k_gemm<8, 4, false, true, true><<<gm, 256, 0, stream>>>(xB, swz + OFF_IPB, ipB_b, hB, N_B);
    k_cvt<<<gc, 256, 0, stream>>>((const float4*)hA, (uint2*)hAbf, NH / 4);
    k_cvt<<<gc, 256, 0, stream>>>((const float4*)hB, (uint2*)hBbf, NH / 4);

    // ---- 4) message-passing layers ----
    int gs = (NE * 32) / 256;       // 62500 blocks
    for (int l = 0; l < 2; ++l) {
        int b = OFF_L0 + l * L_STRIDE;
        // self linear (bias, no relu yet) — bf16-A fast path
        k_gemm<8, 4, true, true, false><<<gm, 256, 0, stream>>>(hAbf, swz + b + 3 * DW_128x128, sAb[l], outA, N_A);
        k_gemm<8, 4, true, true, false><<<gm, 256, 0, stream>>>(hBbf, swz + b + 4 * DW_128x128, sBb[l], outB, N_B);
        // edge ab: (hA @ W_ab) scattered into outB
        k_gemm<8, 4, true, false, false><<<gm, 256, 0, stream>>>(hAbf, swz + b + DW_128x128, nullptr, msg, N_A);
        k_scatter<<<gs, 256, 0, stream>>>(msg, ei_ab, ei_ab + NE, cntAB, outB, NE);
        // edge ba -> outA
        k_gemm<8, 4, true, false, false><<<gm, 256, 0, stream>>>(hBbf, swz + b + 2 * DW_128x128, nullptr, msg, N_B);
        k_scatter<<<gs, 256, 0, stream>>>(msg, ei_ba, ei_ba + NE, cntBA, outA, NE);
        // edge aa -> outA
        k_gemm<8, 4, true, false, false><<<gm, 256, 0, stream>>>(hAbf, swz + b, nullptr, msg, N_A);
        k_scatter<<<gs, 256, 0, stream>>>(msg, ei_aa, ei_aa + NE, cntAA, outA, NE);
        // h = relu(out) + h ; refresh bf16 mirror
        k_combine<<<gc, 256, 0, stream>>>((float4*)hA, (const float4*)outA, (uint2*)hAbf, NH / 4);
        k_combine<<<gc, 256, 0, stream>>>((float4*)hB, (const float4*)outB, (uint2*)hBbf, NH / 4);
    }

    // ---- 5) heads: z = relu(h @ l1W + l1b); pred = z @ l2W + l2b ----
    k_gemm<4, 4, true, true, true><<<gm, 256, 0, stream>>>(hAbf, swz + OFF_HA, hA_l1b, msg, N_A);
    k_head2<<<gn, 256, 0, stream>>>(msg, hA_l2W, hA_l2b, pred, N_A);
    k_gemm<4, 4, true, true, true><<<gm, 256, 0, stream>>>(hBbf, swz + OFF_HB, hB_l1b, msg, N_B);
    k_head2<<<gn, 256, 0, stream>>>(msg, hB_l2W, hB_l2b, pred + N_A, N_B);
}
